// DepthDeformConv_54082228191685
// MI455X (gfx1250) — compile-verified, matrix operand order checked by reading, not run on hardware
//
#include <hip/hip_runtime.h>
#include <stdint.h>

#define KH 3
#define KW 3
#define KK 9
#define PADC 2
#define DILC 2
#define CIN 256
#define COUT 256
#define Hdim 64
#define Wdim 64
#define NSPAT (Hdim * Wdim)        // 4096 per batch image
#define BATCH 4
#define NTOT (BATCH * NSPAT)       // 16384 = GEMM N
#define KDIM (CIN * KK)            // 2304 = GEMM K
#define BK 32
#define BN 64
#define LDT 40                     // padded LDS row stride (ushorts), 80B = 16B aligned

#define USE_ASYNC 1                // CDNA5 global_load_async_to_lds_b128 for A tile

typedef __attribute__((ext_vector_type(16))) __bf16 v16bf;
typedef __attribute__((ext_vector_type(8)))  __bf16 bf16x8;
typedef __attribute__((ext_vector_type(8)))  float  v8f;

__device__ __forceinline__ uint16_t f32_to_bf16(float f) {
    union { float f; uint32_t u; } cv; cv.f = f;
    uint32_t u = cv.u;
    uint32_t r = u + 0x7FFFu + ((u >> 16) & 1u);   // round-to-nearest-even
    return (uint16_t)(r >> 16);
}

__device__ __forceinline__ uint32_t pack2_bf16(float f0, float f1) {
    union { float f; uint32_t u; } a, b; a.f = f0; b.f = f1;
    uint32_t r0 = a.u + 0x7FFFu + ((a.u >> 16) & 1u);
    uint32_t r1 = b.u + 0x7FFFu + ((b.u >> 16) & 1u);
    return (r0 >> 16) | (r1 & 0xFFFF0000u);
}

// ---------------------------------------------------------------------------
// Kernel 1: per-pixel depth offset index: argmin over 3 dilated 3x3 avg pools
// ---------------------------------------------------------------------------
__global__ void ddc_depth_idx(const float* __restrict__ depth,
                              float* __restrict__ idxbuf) {
    int t = blockIdx.x * blockDim.x + threadIdx.x;
    if (t >= NTOT) return;
    int b = t >> 12;
    int s = t & 4095;
    int oh = s >> 6, ow = s & 63;
    const float* dp = depth + b * NSPAT;
    float center = dp[s];
    float sums[3];
#pragma unroll
    for (int d = 1; d <= 3; ++d) {
        float acc = 0.f;
#pragma unroll
        for (int iy = -1; iy <= 1; ++iy) {
            int y = oh + iy * d;
#pragma unroll
            for (int ix = -1; ix <= 1; ++ix) {
                int x = ow + ix * d;
                if (y >= 0 && y < Hdim && x >= 0 && x < Wdim)
                    acc += dp[y * Wdim + x];
            }
        }
        sums[d - 1] = acc;
    }
    float s_m = fabsf(sums[0] * (1.f / 9.f) - center);
    float s_o = fabsf(sums[1] * (1.f / 9.f) - center);
    float s_p = fabsf(sums[2] * (1.f / 9.f) - center);
    int a = 0; float best = s_m;
    if (s_o < best) { best = s_o; a = 1; }
    if (s_p < best) { a = 2; }
    idxbuf[t] = (float)(a - 1);
}

// ---------------------------------------------------------------------------
// Kernel 2: weight (Cout,Cin,3,3) f32 -> bf16 transposed to [o][k*256 + c]
// ---------------------------------------------------------------------------
__global__ void ddc_weight_prep(const float* __restrict__ w,
                                uint16_t* __restrict__ wt) {
    int i = blockIdx.x * blockDim.x + threadIdx.x;
    if (i >= COUT * KDIM) return;
    int o = i / KDIM;
    int rem = i - o * KDIM;
    int k = rem >> 8;
    int c = rem & 255;
    wt[i] = f32_to_bf16(w[(o * CIN + c) * KK + k]);
}

// ---------------------------------------------------------------------------
// Kernel 3: fused deformable-im2col + implicit GEMM with WMMA bf16
//   grid.x = NTOT/BN (256 WGs), 256 threads (8 waves: 4 along M, 2 along N)
//   tap-major K order: coords computed once per tap, reused for 8 c-slabs;
//   double-buffered LDS tiles, one barrier per slab; async A-tile staging;
//   B-tile written as one packed ds_store_b128 per thread per slab.
// ---------------------------------------------------------------------------
__launch_bounds__(256)
__global__ void ddc_main(const float* __restrict__ input,
                         const float* __restrict__ offset,
                         const float* __restrict__ mask,
                         const uint16_t* __restrict__ wt,
                         const float* __restrict__ idxbuf,
                         const float* __restrict__ bias,
                         float* __restrict__ out) {
    __shared__ uint16_t Atile[2][COUT * LDT];   // 2 x 20KB bf16 weights
    __shared__ uint16_t Btile[2][BN * LDT];     // 2 x 5KB  bf16 samples

    const int tid  = threadIdx.x;
    const int lane = tid & 31;
    const int wid  = tid >> 5;
    const int wm   = wid & 3;       // 64-row M block
    const int wn   = wid >> 2;      // 32-col N block

    const int n0 = blockIdx.x * BN;
    const int b  = n0 >> 12;        // BN=64 divides 4096: whole WG in one batch
    const int s0 = n0 & 4095;

    // B-tile build assignment: thread -> (n = tid%64, 8 consecutive channels)
    const int bn  = tid & 63;
    const int bc8 = (tid >> 6) * 8;  // channel start within slab: 0,8,16,24
    const int s   = s0 + bn;
    const int oh  = s >> 6, ow = s & 63;
    const float idxv = idxbuf[n0 + bn];

    // A-tile copy assignment: thread -> 4 rows x 16B
    const int acol8 = (tid & 3) * 8;
    const int arow0 = tid >> 2;

    v8f acc[4][2];
#pragma unroll
    for (int i = 0; i < 4; ++i)
#pragma unroll
        for (int j = 0; j < 2; ++j)
#pragma unroll
            for (int e = 0; e < 8; ++e) acc[i][j][e] = 0.f;

    const float* inpB = input  + (size_t)b * CIN * NSPAT;
    const float* offB = offset + (size_t)b * 2 * KK * NSPAT;
    const float* mskB = mask   + (size_t)b * KK * NSPAT;

    const int r    = lane & 15;
    const int koff = (lane >> 4) * 8;   // ISA 16-bit frag K-chunk layout

    for (int k = 0; k < KK; ++k) {
        // ---- deformable coords: depend on (n, k) only; hoisted over c-slabs
        int kr = k / 3, kc = k - kr * 3;
        float offy = offB[(2 * k) * NSPAT + s];
        float offx = offB[(2 * k + 1) * NSPAT + s];
        float m    = mskB[k * NSPAT + s];
        float ys = (float)(oh - PADC + kr * DILC) + idxv * (float)(kr - 1) + offy;
        float xs = (float)(ow - PADC + kc * DILC) + idxv * (float)(kc - 1) + offx;
        float y0f = floorf(ys), x0f = floorf(xs);
        int y0 = (int)y0f, x0 = (int)x0f;
        int y1 = y0 + 1,   x1 = x0 + 1;
        float wy = ys - y0f, wx = xs - x0f;
        float vy0 = (y0 >= 0 && y0 < Hdim) ? 1.f : 0.f;
        float vy1 = (y1 >= 0 && y1 < Hdim) ? 1.f : 0.f;
        float vx0 = (x0 >= 0 && x0 < Wdim) ? 1.f : 0.f;
        float vx1 = (x1 >= 0 && x1 < Wdim) ? 1.f : 0.f;
        int y0c = min(max(y0, 0), Hdim - 1), y1c = min(max(y1, 0), Hdim - 1);
        int x0c = min(max(x0, 0), Wdim - 1), x1c = min(max(x1, 0), Wdim - 1);
        float w00 = (1.f - wy) * (1.f - wx) * vy0 * vx0 * m;
        float w01 = (1.f - wy) * wx         * vy0 * vx1 * m;
        float w10 = wy         * (1.f - wx) * vy1 * vx0 * m;
        float w11 = wy         * wx         * vy1 * vx1 * m;
        int a00 = y0c * Wdim + x0c, a01 = y0c * Wdim + x1c;
        int a10 = y1c * Wdim + x0c, a11 = y1c * Wdim + x1c;

        for (int cs = 0; cs < 8; ++cs) {
            const int c0  = cs << 5;
            const int kk0 = (k << 8) + c0;
            const int buf = cs & 1;           // k*8 is even -> slab parity == cs parity
            uint16_t* At = Atile[buf];
            uint16_t* Bt = Btile[buf];

            // ---- A tile: 256 rows x 32 bf16, async DMA into LDS ----
#if USE_ASYNC
#pragma unroll
            for (int p = 0; p < 4; ++p) {
                int row = arow0 + p * 64;
                uint64_t ga = (uint64_t)(uintptr_t)(wt + (size_t)row * KDIM + kk0 + acol8);
                uint32_t la = (uint32_t)(uintptr_t)&At[row * LDT + acol8]; // LDS offset = addr[31:0]
                asm volatile("global_load_async_to_lds_b128 %0, %1, off"
                             :: "v"(la), "v"(ga) : "memory");
            }
#else
#pragma unroll
            for (int p = 0; p < 4; ++p) {
                int row = arow0 + p * 64;
                uint4 v = *(const uint4*)(wt + (size_t)row * KDIM + kk0 + acol8);
                *(uint4*)&At[row * LDT + acol8] = v;
            }
#endif

            // ---- B tile: bilinear gathers over 8 consecutive channels,
            //      packed into one ds_store_b128 per thread ----
            {
                const float* chan = inpB + (size_t)(c0 + bc8) * NSPAT;
                float v[8];
#pragma unroll
                for (int j = 0; j < 8; ++j) {
                    v[j] = w00 * chan[a00] + w01 * chan[a01]
                         + w10 * chan[a10] + w11 * chan[a11];
                    chan += NSPAT;
                }
                uint4 pk;
                pk.x = pack2_bf16(v[0], v[1]);
                pk.y = pack2_bf16(v[2], v[3]);
                pk.z = pack2_bf16(v[4], v[5]);
                pk.w = pack2_bf16(v[6], v[7]);
                *(uint4*)&Bt[bn * LDT + bc8] = pk;   // bn*80 + bc8*2: 16B aligned
            }

#if USE_ASYNC
            asm volatile("s_wait_asynccnt 0x0" ::: "memory");
#endif
            __syncthreads();   // single barrier per slab (double-buffered tiles)

            // ---- WMMA: 4x2 tiles of 16x16, one K=32 step per slab ----
            {
                v16bf bfrag[2];
#pragma unroll
                for (int j = 0; j < 2; ++j) {
                    const uint16_t* bp = &Bt[(wn * 32 + j * 16 + r) * LDT + koff];
                    bf16x8 lo = *(const bf16x8*)bp;
                    bf16x8 hi = *(const bf16x8*)(bp + 16);
#pragma unroll
                    for (int e = 0; e < 8; ++e) { bfrag[j][e] = lo[e]; bfrag[j][8 + e] = hi[e]; }
                }
#pragma unroll
                for (int i = 0; i < 4; ++i) {
                    const uint16_t* ap = &At[(wm * 64 + i * 16 + r) * LDT + koff];
                    bf16x8 lo = *(const bf16x8*)ap;
                    bf16x8 hi = *(const bf16x8*)(ap + 16);
                    v16bf afrag;
#pragma unroll
                    for (int e = 0; e < 8; ++e) { afrag[e] = lo[e]; afrag[8 + e] = hi[e]; }
#pragma unroll
                    for (int j = 0; j < 2; ++j) {
                        acc[i][j] = __builtin_amdgcn_wmma_f32_16x16x32_bf16(
                            false, afrag, false, bfrag[j], (short)0, acc[i][j],
                            false, false);
                    }
                }
            }
        }
    }

    // ---- epilogue: C/D layout -> VGPR e: M = e + (lane>>4)*8, N = lane&15 ----
    const int cn  = lane & 15;
    const int cm8 = (lane >> 4) * 8;
#pragma unroll
    for (int i = 0; i < 4; ++i) {
#pragma unroll
        for (int j = 0; j < 2; ++j) {
            int n_glob = n0 + wn * 32 + j * 16 + cn;
            int sg = n_glob & 4095;
            float* op = out + (size_t)b * COUT * NSPAT + sg;
#pragma unroll
            for (int e = 0; e < 8; ++e) {
                int o = wm * 64 + i * 16 + e + cm8;
                op[(size_t)o * NSPAT] = acc[i][j][e] + bias[o];
            }
        }
    }
}

// ---------------------------------------------------------------------------
extern "C" void kernel_launch(void* const* d_in, const int* in_sizes, int n_in,
                              void* d_out, int out_size, void* d_ws, size_t ws_size,
                              hipStream_t stream) {
    (void)in_sizes; (void)n_in; (void)out_size; (void)ws_size;
    const float* input  = (const float*)d_in[0];
    const float* depth  = (const float*)d_in[1];
    const float* offset = (const float*)d_in[2];
    const float* mask   = (const float*)d_in[3];
    const float* weight = (const float*)d_in[4];
    const float* bias   = (const float*)d_in[5];
    float* out = (float*)d_out;

    uint16_t* wt  = (uint16_t*)d_ws;                                      // 1.18 MB
    float* idxbuf = (float*)((char*)d_ws + (size_t)COUT * KDIM * sizeof(uint16_t)); // 64 KB

    ddc_depth_idx<<<(NTOT + 255) / 256, 256, 0, stream>>>(depth, idxbuf);
    ddc_weight_prep<<<(COUT * KDIM + 255) / 256, 256, 0, stream>>>(weight, wt);
    ddc_main<<<NTOT / BN, 256, 0, stream>>>(input, offset, mask, wt, idxbuf, bias, out);
}
